// UltrametricCausalSelfAttention_55654186221823
// MI455X (gfx1250) — matrix operand order repaired and checked by reading, not compiled, not run on hardware
//
#include <hip/hip_runtime.h>
#include <hip/hip_bf16.h>

typedef __attribute__((ext_vector_type(16))) _Float16 v16h;
typedef __attribute__((ext_vector_type(8)))  _Float16 v8h;
typedef __attribute__((ext_vector_type(8)))  float    v8f;

static constexpr int kB = 2;
static constexpr int kT = 512;
static constexpr int kC = 1024;
static constexpr int kH = 16;
static constexpr int kD = 64;      // head dim
static constexpr int kDH = 32;     // half head dim (rope split)
static constexpr int kKDIG = 8;
static constexpr int kBT = kB * kT;   // 1024
static constexpr float kBeta = 32.0f; // LCP_BETA

#define WMMA_F16(A_, B_, C_) \
  __builtin_amdgcn_wmma_f32_16x16x32_f16(false, (A_), false, (B_), (short)0, (C_), false, false)

// ---------------------------------------------------------------------------
// Elementwise f32 -> f16 (8 elements / thread, vectorizable)
// ---------------------------------------------------------------------------
__global__ void cvt_f32_f16(const float* __restrict__ src, _Float16* __restrict__ dst, int n) {
  const int i = (blockIdx.x * blockDim.x + threadIdx.x) * 8;
  if (i >= n) return;
#pragma unroll
  for (int j = 0; j < 8; ++j) dst[i + j] = (_Float16)src[i + j];
}

// ---------------------------------------------------------------------------
// W[K,N] f32 -> WT[N,K] f16, LDS-tiled 32x32 transpose (coalesced both ways)
// ---------------------------------------------------------------------------
__global__ void transpose_cvt_f32_f16(const float* __restrict__ W, _Float16* __restrict__ WT,
                                      int K, int N) {
  __shared__ _Float16 tile[32][33];
  const int k0 = blockIdx.y * 32;
  const int n0 = blockIdx.x * 32;
  const int tx = threadIdx.x, ty = threadIdx.y;    // (32, 8)
#pragma unroll
  for (int i = 0; i < 4; ++i)
    tile[ty + i * 8][tx] = (_Float16)W[(size_t)(k0 + ty + i * 8) * N + n0 + tx];
  __syncthreads();
#pragma unroll
  for (int i = 0; i < 4; ++i)
    WT[(size_t)(n0 + ty + i * 8) * K + k0 + tx] = tile[tx][ty + i * 8];
}

// ---------------------------------------------------------------------------
// Batched f16 transpose: src[batch][R][Cc] -> dst[batch][Cc][R]
// ---------------------------------------------------------------------------
__global__ void transpose_h16(const _Float16* __restrict__ S, _Float16* __restrict__ Dp,
                              int R, int Cc) {
  __shared__ _Float16 tile[32][33];
  const int batch = blockIdx.z;
  const _Float16* s = S + (size_t)batch * R * Cc;
  _Float16* d = Dp + (size_t)batch * R * Cc;
  const int r0 = blockIdx.y * 32;
  const int c0 = blockIdx.x * 32;
  const int tx = threadIdx.x, ty = threadIdx.y;    // (32, 8)
#pragma unroll
  for (int i = 0; i < 4; ++i)
    tile[ty + i * 8][tx] = s[(size_t)(r0 + ty + i * 8) * Cc + c0 + tx];
  __syncthreads();
#pragma unroll
  for (int i = 0; i < 4; ++i)
    d[(size_t)(c0 + ty + i * 8) * R + r0 + tx] = tile[tx][ty + i * 8];
}

// ---------------------------------------------------------------------------
// GEMM: C[M,N] = A[M,K] @ B[K,N] with A f16 row-major, B given TRANSPOSED
// (BT[N,K] f16). One wave -> 16x64 strip; 8 waves/block share the col strip.
// All fragment loads are contiguous 16/32-byte vector loads.
// ---------------------------------------------------------------------------
__global__ void gemm_f16f16(const _Float16* __restrict__ A, const _Float16* __restrict__ BT,
                            float* __restrict__ Cm, int M, int N, int K) {
  const int lane = threadIdx.x & 31;
  const int wib  = threadIdx.x >> 5;               // wave in block (0..7)
  const int nColStrips = N >> 6;
  const int blockCol = blockIdx.x % nColStrips;
  const int blockRow = blockIdx.x / nColStrips;
  const int row = blockRow * 128 + wib * 16;
  const int colBase = blockCol << 6;
  if (row >= M) return;

  const int m     = row + (lane & 15);
  const int khalf = lane >> 4;
  const int n0    = colBase + (lane & 15);

  const _Float16* Arow  = A + (size_t)m * K;
  const _Float16* Brow0 = BT + (size_t)(n0)      * K;
  const _Float16* Brow1 = BT + (size_t)(n0 + 16) * K;
  const _Float16* Brow2 = BT + (size_t)(n0 + 32) * K;
  const _Float16* Brow3 = BT + (size_t)(n0 + 48) * K;

  v8f acc0 = {}, acc1 = {}, acc2 = {}, acc3 = {};

  for (int kb = 0; kb < K; kb += 32) {
    // A fragment: K pattern {khalf*16+0..7, khalf*16+8..15} = two 16B chunks
    const v8h alo = *(const v8h*)(Arow + kb + (khalf << 4));
    const v8h ahi = *(const v8h*)(Arow + kb + 8 + (khalf << 4));
    const v16h a = __builtin_shufflevector(alo, ahi,
        0, 1, 2, 3, 4, 5, 6, 7, 8, 9, 10, 11, 12, 13, 14, 15);

    // B fragments: contiguous 16 halves of BT row (K = khalf*16 + 0..15)
    const int kofs = kb + (khalf << 4);
    const v16h b0 = *(const v16h*)(Brow0 + kofs);
    const v16h b1 = *(const v16h*)(Brow1 + kofs);
    const v16h b2 = *(const v16h*)(Brow2 + kofs);
    const v16h b3 = *(const v16h*)(Brow3 + kofs);

    if (kb + 32 < K) {
      __builtin_prefetch(Arow + kb + 32, 0, 1);     // global_prefetch_b8
      __builtin_prefetch(Brow0 + kofs + 32, 0, 1);
    }

    acc0 = WMMA_F16(a, b0, acc0);
    acc1 = WMMA_F16(a, b1, acc1);
    acc2 = WMMA_F16(a, b2, acc2);
    acc3 = WMMA_F16(a, b3, acc3);
  }

  // C layout: VGPR r -> M = r (lanes 0-15) / r+8 (lanes 16-31); N = lane&15
  const int rowHi = (lane >> 4) << 3;
  const int cn = colBase + (lane & 15);
#pragma unroll
  for (int r = 0; r < 8; ++r) {
    const int mo = row + rowHi + r;
    Cm[(size_t)mo * N + cn]      = acc0[r];
    Cm[(size_t)mo * N + cn + 16] = acc1[r];
    Cm[(size_t)mo * N + cn + 32] = acc2[r];
    Cm[(size_t)mo * N + cn + 48] = acc3[r];
  }
}

// ---------------------------------------------------------------------------
// RoPE + RMS-norm -> digit projections (sigmoid) + V pack to f16 [B,H,T,D].
// One wave per (b,t,h); lane l owns the RoPE pair (l, l+32).
// (Normalized q/k themselves are dead beyond the digits -> not stored.)
// ---------------------------------------------------------------------------
__global__ void rope_rms_digits(const float* __restrict__ qraw,
                                const float* __restrict__ kraw,
                                const float* __restrict__ vraw,
                                const float* __restrict__ cosT,
                                const float* __restrict__ sinT,
                                const float* __restrict__ Wdq,
                                const float* __restrict__ Wdk,
                                _Float16* __restrict__ vh,
                                float* __restrict__ qdig,
                                float* __restrict__ kdig) {
  const int wave = (blockIdx.x * blockDim.x + threadIdx.x) >> 5;
  const int lane = threadIdx.x & 31;
  const int h = wave % kH;
  const int bt = wave / kH;
  const int t = bt % kT;
  const int b = bt / kT;

  const float c = cosT[t * kDH + lane];
  const float s = sinT[t * kDH + lane];
  const size_t inOff  = ((size_t)(b * kT + t) * kC) + h * kD;
  const size_t outOff = (((size_t)(b * kH + h) * kT) + t) * kD;
  const size_t digOff = (((size_t)(b * kH + h) * kT) + t) * kKDIG;

  // ---- Q digits ----
  {
    const float x1 = qraw[inOff + lane];
    const float x2 = qraw[inOff + lane + kDH];
    float r1 = x1 * c - x2 * s;
    float r2 = x1 * s + x2 * c;
    float ss = r1 * r1 + r2 * r2;
#pragma unroll
    for (int off = 16; off; off >>= 1) ss += __shfl_xor(ss, off, 32);
    const float scale = rsqrtf(ss * (1.0f / kD) + 1e-6f);
    r1 *= scale; r2 *= scale;
#pragma unroll
    for (int j = 0; j < kKDIG; ++j) {
      float p = r1 * Wdq[lane * kKDIG + j] + r2 * Wdq[(lane + kDH) * kKDIG + j];
#pragma unroll
      for (int off = 16; off; off >>= 1) p += __shfl_xor(p, off, 32);
      if (lane == 0) qdig[digOff + j] = 1.0f / (1.0f + __expf(-p));
    }
  }

  // ---- K digits ----
  {
    const float x1 = kraw[inOff + lane];
    const float x2 = kraw[inOff + lane + kDH];
    float r1 = x1 * c - x2 * s;
    float r2 = x1 * s + x2 * c;
    float ss = r1 * r1 + r2 * r2;
#pragma unroll
    for (int off = 16; off; off >>= 1) ss += __shfl_xor(ss, off, 32);
    const float scale = rsqrtf(ss * (1.0f / kD) + 1e-6f);
    r1 *= scale; r2 *= scale;
#pragma unroll
    for (int j = 0; j < kKDIG; ++j) {
      float p = r1 * Wdk[lane * kKDIG + j] + r2 * Wdk[(lane + kDH) * kKDIG + j];
#pragma unroll
      for (int off = 16; off; off >>= 1) p += __shfl_xor(p, off, 32);
      if (lane == 0) kdig[digOff + j] = 1.0f / (1.0f + __expf(-p));
    }
  }

  // ---- V: pack to f16 [B,H,T,D] (transposed to [B,H,D,T] by a later pass) ----
  vh[outOff + lane]       = (_Float16)vraw[inOff + lane];
  vh[outOff + lane + kDH] = (_Float16)vraw[inOff + lane + kDH];
}

// ---------------------------------------------------------------------------
// Ultrametric causal attention. One wave per (b, h, 16-row q-tile).
// Lane computes exactly its 16 A-fragment weights in-register; V read from the
// transposed copy so B fragments are contiguous 32-byte loads. Writes y in f16
// (it is the A operand of the final projection GEMM).
// ---------------------------------------------------------------------------
__global__ void attn_wmma(const float* __restrict__ qdig,
                          const float* __restrict__ kdig,
                          const _Float16* __restrict__ vT,   // [B,H,D,T]
                          _Float16* __restrict__ y16) {      // [B,T,C]
  const int wave = (blockIdx.x * blockDim.x + threadIdx.x) >> 5;
  const int lane = threadIdx.x & 31;
  const int nqt = kT / 16;
  const int qt = wave % nqt;
  const int bh = wave / nqt;
  const int h = bh % kH;
  const int b = bh / kH;

  const int m = lane & 15;
  const int khalf = lane >> 4;
  const int qrow = qt * 16 + m;
  const int n0 = lane & 15;

  const size_t bhOff = (size_t)(b * kH + h) * kT;
  const float* qdp = qdig + (bhOff + qrow) * kKDIG;
  float qd[kKDIG];
#pragma unroll
  for (int j = 0; j < kKDIG; ++j) qd[j] = qdp[j];

  const float* kdg = kdig + bhOff * kKDIG;
  const _Float16* vt0 = vT + bhOff * kD + (size_t)(n0)      * kT;
  const _Float16* vt1 = vT + bhOff * kD + (size_t)(n0 + 16) * kT;
  const _Float16* vt2 = vT + bhOff * kD + (size_t)(n0 + 32) * kT;
  const _Float16* vt3 = vT + bhOff * kD + (size_t)(n0 + 48) * kT;

  v8f acc0 = {}, acc1 = {}, acc2 = {}, acc3 = {};
  float denom = 0.0f;

  const int nkb = (qt * 16 + 16 + 31) >> 5;       // causal: k in [0, qt*16+15]
  for (int kb0 = 0; kb0 < nkb; ++kb0) {
    const int kb = kb0 * 32;

    // A fragment = 16x32 ultrametric weight tile, computed in-register.
    v16h a;
#pragma unroll
    for (int j = 0; j < 8; ++j) {
      const int kk0 = kb + (j << 1) + (khalf << 4);
#pragma unroll
      for (int e = 0; e < 2; ++e) {
        const int kidx = kk0 + e;
        float w = 0.0f;
        if (kidx <= qrow) {
          const float* kd = kdg + (size_t)kidx * kKDIG;
          float s = 0.0f, lcp = 0.0f;
#pragma unroll
          for (int dg = 0; dg < kKDIG; ++dg) {      // prefix sums -> independent exps
            const float d = qd[dg] - kd[dg];
            s += d * d;
            lcp += __expf(-kBeta * s);
          }
          w = exp2f(lcp);                           // exp(lcp * ln 2)
          denom += w;
        }
        a[2 * j + e] = (_Float16)w;
      }
    }

    // B fragments: contiguous 16 halves of each vT row (K = kb + khalf*16 ..)
    const int krow = kb + (khalf << 4);
    const v16h b0 = *(const v16h*)(vt0 + krow);
    const v16h b1 = *(const v16h*)(vt1 + krow);
    const v16h b2 = *(const v16h*)(vt2 + krow);
    const v16h b3 = *(const v16h*)(vt3 + krow);

    acc0 = WMMA_F16(a, b0, acc0);
    acc1 = WMMA_F16(a, b1, acc1);
    acc2 = WMMA_F16(a, b2, acc2);
    acc3 = WMMA_F16(a, b3, acc3);
  }

  // Fold the two disjoint k-partitions; lane L holds denom for row L&15.
  denom += __shfl_xor(denom, 16, 32);
  denom = fmaxf(denom, 1e-9f);

  const int rowHi = (lane >> 4) << 3;
#pragma unroll
  for (int r = 0; r < 8; ++r) {
    const int trow = qt * 16 + rowHi + r;
    const float inv = 1.0f / __shfl(denom, rowHi + r, 32);
    _Float16* yp = y16 + ((size_t)(b * kT + trow) * kC) + h * kD;
    yp[n0]      = (_Float16)(acc0[r] * inv);
    yp[n0 + 16] = (_Float16)(acc1[r] * inv);
    yp[n0 + 32] = (_Float16)(acc2[r] * inv);
    yp[n0 + 48] = (_Float16)(acc3[r] * inv);
  }
}

// ---------------------------------------------------------------------------
extern "C" void kernel_launch(void* const* d_in, const int* in_sizes, int n_in,
                              void* d_out, int out_size, void* d_ws, size_t ws_size,
                              hipStream_t stream) {
  const float* x    = (const float*)d_in[0];
  const float* cosT = (const float*)d_in[1];
  const float* sinT = (const float*)d_in[2];
  const float* Wq   = (const float*)d_in[3];
  const float* Wk   = (const float*)d_in[4];
  const float* Wv   = (const float*)d_in[5];
  const float* Wo   = (const float*)d_in[6];
  const float* Wdq  = (const float*)d_in[7];
  const float* Wdk  = (const float*)d_in[8];
  float* out = (float*)d_out;

  // Workspace layout
  char* ws = (char*)d_ws;
  const size_t fmat = (size_t)kBT * kC * sizeof(float);             // 4 MB
  const size_t hmat = (size_t)kBT * kC * sizeof(_Float16);          // 2 MB
  const size_t hv   = (size_t)kB * kH * kT * kD * sizeof(_Float16); // 2 MB
  const size_t dmat = (size_t)kB * kH * kT * kKDIG * sizeof(float); // 512 KB
  _Float16* x16  = (_Float16*)ws; ws += hmat;
  _Float16* WqT  = (_Float16*)ws; ws += hmat;
  _Float16* WkT  = (_Float16*)ws; ws += hmat;
  _Float16* WvT  = (_Float16*)ws; ws += hmat;
  _Float16* WoT  = (_Float16*)ws; ws += hmat;
  _Float16* y16  = (_Float16*)ws; ws += hmat;
  float*    qraw = (float*)ws;    ws += fmat;
  float*    kraw = (float*)ws;    ws += fmat;
  float*    vraw = (float*)ws;    ws += fmat;
  _Float16* vh   = (_Float16*)ws; ws += hv;
  _Float16* vTb  = (_Float16*)ws; ws += hv;
  float*    qdg  = (float*)ws;    ws += dmat;
  float*    kdg  = (float*)ws;    ws += dmat;

  // 0) One-time f16 prep: x -> f16, weights -> transposed f16
  const int nElem = kBT * kC;
  cvt_f32_f16<<<nElem / (8 * 256), 256, 0, stream>>>(x, x16, nElem);
  {
    dim3 tg(32, 8), tb(kC / 32, kC / 32);
    transpose_cvt_f32_f16<<<tb, tg, 0, stream>>>(Wq, WqT, kC, kC);
    transpose_cvt_f32_f16<<<tb, tg, 0, stream>>>(Wk, WkT, kC, kC);
    transpose_cvt_f32_f16<<<tb, tg, 0, stream>>>(Wv, WvT, kC, kC);
    transpose_cvt_f32_f16<<<tb, tg, 0, stream>>>(Wo, WoT, kC, kC);
  }

  // 1) QKV projections (f16 WMMA, f32 accumulate/output)
  const int gemmBlocks = (kBT / 128) * (kC / 64);   // 8 * 16 = 128
  gemm_f16f16<<<gemmBlocks, 256, 0, stream>>>(x16, WqT, qraw, kBT, kC, kC);
  gemm_f16f16<<<gemmBlocks, 256, 0, stream>>>(x16, WkT, kraw, kBT, kC, kC);
  gemm_f16f16<<<gemmBlocks, 256, 0, stream>>>(x16, WvT, vraw, kBT, kC, kC);

  // 2) RoPE + RMS-norm + digit projection; V pack to f16
  const int rBlocks = (kB * kT * kH) * 32 / 256;    // 2048
  rope_rms_digits<<<rBlocks, 256, 0, stream>>>(qraw, kraw, vraw, cosT, sinT,
                                               Wdq, Wdk, vh, qdg, kdg);

  // 2b) V -> [B,H,D,T] so attention B-fragments are contiguous
  {
    dim3 tg(32, 8), tb(kD / 32, kT / 32, kB * kH);
    transpose_h16<<<tb, tg, 0, stream>>>(vh, vTb, kT, kD);
  }

  // 3) Ultrametric causal attention (WMMA PV product, fused normalization)
  const int aBlocks = (kB * kH * (kT / 16)) * 32 / 256;  // 128
  attn_wmma<<<aBlocks, 256, 0, stream>>>(qdg, kdg, vTb, y16);

  // 4) Output projection
  gemm_f16f16<<<gemmBlocks, 256, 0, stream>>>(y16, WoT, out, kBT, kC, kC);
}
